// RNN_26980984553685
// MI455X (gfx1250) — matrix operand order repaired and checked by reading, not compiled, not run on hardware
//
#include <hip/hip_runtime.h>
#include <hip/hip_bf16.h>

typedef __bf16 bf16;
typedef __attribute__((ext_vector_type(16))) __bf16 v16bf;
typedef __attribute__((ext_vector_type(8)))  __bf16 v8bf;
typedef __attribute__((ext_vector_type(8)))  float  v8f;

#define kB  32
#define kT  512
#define kE  768
#define kH  3132
#define kO  19
#define kHp 3136   // H padded to mult of 32 (K) / 16 (N)
#define kEp 768    // already mult of 32
#define kOp 32     // O padded to mult of 16

// ---------------- device helpers ----------------

// 16-bit A-matrix 16x32 fragment (ISA 7.12.2): lane<16 holds K {kb..kb+7, kb+16..kb+23},
// lane>=16 holds K {kb+8..kb+15, kb+24..kb+31}; row M = lane&15.
__device__ inline v16bf load_fragA(const bf16* __restrict__ rowp, int kb, int kh) {
  const v8bf lo = *(const v8bf*)(rowp + kb + kh * 8);
  const v8bf hi = *(const v8bf*)(rowp + kb + 16 + kh * 8);
  v16bf r;
#pragma unroll
  for (int i = 0; i < 8; ++i) { r[i] = lo[i]; r[i + 8] = hi[i]; }
  return r;
}

// B-matrix 32x16 fragment: lane<16 holds K {kb..kb+7, kb+16..kb+23} of column n=lane, lane>=16 holds
// K {kb+8..kb+15, kb+24..kb+31} of column n=lane-16 (same K order as A).
__device__ inline v16bf load_fragB(const bf16* __restrict__ colp, int kb, int kh) {
  const v8bf lo = *(const v8bf*)(colp + kb + kh * 8);
  const v8bf hi = *(const v8bf*)(colp + kb + 16 + kh * 8);
  v16bf r;
#pragma unroll
  for (int i = 0; i < 8; ++i) { r[i] = lo[i]; r[i + 8] = hi[i]; }
  return r;
}

__device__ inline float block_sum256(float v, float* sbuf) {
#pragma unroll
  for (int m = 16; m >= 1; m >>= 1) v += __shfl_xor(v, m, 32);
  const int tid = threadIdx.x;
  if ((tid & 31) == 0) sbuf[tid >> 5] = v;
  __syncthreads();
  float r = (tid < 8) ? sbuf[tid] : 0.f;
  if (tid < 32) {
#pragma unroll
    for (int m = 4; m >= 1; m >>= 1) r += __shfl_xor(r, m, 32);
    if (tid == 0) sbuf[0] = r;
  }
  __syncthreads();
  float out = sbuf[0];
  __syncthreads();
  return out;
}

__device__ inline float block_max256(float v, float* sbuf) {
#pragma unroll
  for (int m = 16; m >= 1; m >>= 1) v = fmaxf(v, __shfl_xor(v, m, 32));
  const int tid = threadIdx.x;
  if ((tid & 31) == 0) sbuf[tid >> 5] = v;
  __syncthreads();
  float r = (tid < 8) ? sbuf[tid] : -1e30f;
  if (tid < 32) {
#pragma unroll
    for (int m = 4; m >= 1; m >>= 1) r = fmaxf(r, __shfl_xor(r, m, 32));
    if (tid == 0) sbuf[0] = r;
  }
  __syncthreads();
  float out = sbuf[0];
  __syncthreads();
  return out;
}

// ---------------- generic 2-segment WMMA GEMM:  D = tanh?( A0@B0^T + A1@B1^T + Cin + bias ) ----
// A*: [M, K*p] bf16 row-major (zero-padded K tail). B*: [Np, K*p] bf16 row-major (weights, "W^T" form).
// ep: 0 = f32 store (col<Nreal only), 1 = bf16 store (0 in pad cols), 2 = tanh+bf16 (+opt dup),
//     3 = tanh + f32 store + per-row sum/sumsq atomics (for LayerNorm).
__global__ __launch_bounds__(128)
void gemm_wmma_kernel(const bf16* __restrict__ A0, int lda0, int K0,
                      const bf16* __restrict__ B0, int ldb0,
                      const bf16* __restrict__ A1, int lda1, int K1,
                      const bf16* __restrict__ B1, int ldb1,
                      const bf16* __restrict__ Cin, int ldcin,
                      const float* __restrict__ bias0, const float* __restrict__ bias1,
                      int Np, int Nreal, int ep,
                      float* __restrict__ Df, bf16* __restrict__ Db, bf16* __restrict__ Db2,
                      int ldd, float* __restrict__ stats)
{
  const int wave = threadIdx.x >> 5;
  const int lane = threadIdx.x & 31;
  const int sub  = lane & 15;
  const int kh   = lane >> 4;
  const int n0   = (blockIdx.x * 4 + wave) * 16;
  if (n0 >= Np) return;                 // wave-uniform exit: EXEC stays all-ones for WMMA
  const int m0  = blockIdx.y * 32;
  const int col = n0 + sub;

  v8f acc0 = {0.f,0.f,0.f,0.f,0.f,0.f,0.f,0.f};
  v8f acc1 = {0.f,0.f,0.f,0.f,0.f,0.f,0.f,0.f};

  {
    const bf16* a0p = A0 + (size_t)(m0 + sub)      * lda0;
    const bf16* a1p = A0 + (size_t)(m0 + 16 + sub) * lda0;
    const bf16* bp  = B0 + (size_t)col * ldb0;
    for (int kb = 0; kb < K0; kb += 32) {
      v16bf bfr = load_fragB(bp,  kb, kh);
      v16bf af0 = load_fragA(a0p, kb, kh);
      v16bf af1 = load_fragA(a1p, kb, kh);
      acc0 = __builtin_amdgcn_wmma_f32_16x16x32_bf16(false, af0, false, bfr, (short)0, acc0, false, false);
      acc1 = __builtin_amdgcn_wmma_f32_16x16x32_bf16(false, af1, false, bfr, (short)0, acc1, false, false);
    }
  }
  if (A1) {
    const bf16* a0p = A1 + (size_t)(m0 + sub)      * lda1;
    const bf16* a1p = A1 + (size_t)(m0 + 16 + sub) * lda1;
    const bf16* bp  = B1 + (size_t)col * ldb1;
    for (int kb = 0; kb < K1; kb += 32) {
      v16bf bfr = load_fragB(bp,  kb, kh);
      v16bf af0 = load_fragA(a0p, kb, kh);
      v16bf af1 = load_fragA(a1p, kb, kh);
      acc0 = __builtin_amdgcn_wmma_f32_16x16x32_bf16(false, af0, false, bfr, (short)0, acc0, false, false);
      acc1 = __builtin_amdgcn_wmma_f32_16x16x32_bf16(false, af1, false, bfr, (short)0, acc1, false, false);
    }
  }

  const bool valid = (col < Nreal);
  float bias = 0.f;
  if (valid) {
    if (bias0) bias += bias0[col];
    if (bias1) bias += bias1[col];
  }

#pragma unroll
  for (int mt = 0; mt < 2; ++mt) {
    v8f acc = mt ? acc1 : acc0;
#pragma unroll
    for (int i = 0; i < 8; ++i) {
      const int row = m0 + mt * 16 + kh * 8 + i;   // D layout: vgpr i -> M=i (+8 for upper half)
      float v = acc[i] + bias;
      if (Cin) v += (float)Cin[(size_t)row * ldcin + col];
      if (!valid) v = 0.f;
      if (ep == 0) {
        if (valid) Df[(size_t)row * ldd + col] = v;
      } else if (ep == 1) {
        Db[(size_t)row * ldd + col] = (bf16)v;
      } else if (ep == 2) {
        v = tanhf(v);
        Db[(size_t)row * ldd + col] = (bf16)v;
        if (Db2) Db2[(size_t)row * ldd + col] = (bf16)v;
      } else {                                     // ep == 3 : tanh + stats for LN
        v = tanhf(v);
        Df[(size_t)row * ldd + col] = v;
        float s = v, s2 = v * v;
#pragma unroll
        for (int m = 1; m < 16; m <<= 1) {         // reduce across 16 cols of same row
          s  += __shfl_xor(s,  m, 32);
          s2 += __shfl_xor(s2, m, 32);
        }
        if (sub == 0) {
          atomicAdd(&stats[row * 2 + 0], s);
          atomicAdd(&stats[row * 2 + 1], s2);
        }
      }
    }
  }
}

// ---------------- elementwise / reduction kernels ----------------

// Input LayerNorm over E, writing bf16 rows ordered r = t*B + b.
__global__ __launch_bounds__(256)
void input_ln_kernel(const float* __restrict__ x, const float* __restrict__ g,
                     const float* __restrict__ bta, bf16* __restrict__ xn)
{
  __shared__ float sbuf[8];
  const int r = blockIdx.x;
  const int t = r / kB, b = r % kB;
  const float* xr = x + ((size_t)b * kT + t) * kE;
  float s = 0.f, s2 = 0.f;
  for (int e = threadIdx.x; e < kE; e += 256) { float v = xr[e]; s += v; s2 += v * v; }
  s  = block_sum256(s,  sbuf);
  s2 = block_sum256(s2, sbuf);
  const float mean = s / kE;
  const float rstd = rsqrtf(s2 / kE - mean * mean + 1e-5f);
  bf16* outr = xn + (size_t)r * kEp;
  for (int e = threadIdx.x; e < kE; e += 256)
    outr[e] = (bf16)((xr[e] - mean) * rstd * g[e] + bta[e]);
}

// Apply LayerNorm to tanh outputs using accumulated sum/sumsq; zero pad cols.
__global__ void ln_apply_kernel(const float* __restrict__ t1, const float* __restrict__ acc,
                                const float* __restrict__ g, const float* __restrict__ bta,
                                bf16* __restrict__ out)
{
  const int idx = blockIdx.x * blockDim.x + threadIdx.x;
  if (idx >= kB * kHp) return;
  const int b = idx / kHp, n = idx - b * kHp;
  float v = 0.f;
  if (n < kH) {
    const float mean = acc[b * 2] / kH;
    const float rstd = rsqrtf(acc[b * 2 + 1] / kH - mean * mean + 1e-5f);
    v = (t1[idx] - mean) * rstd * g[n] + bta[n];
  }
  out[idx] = (bf16)v;
}

__global__ __launch_bounds__(256)
void dot_qbias_kernel(const bf16* __restrict__ q, const float* __restrict__ bk,
                      int D, int Dp, float* __restrict__ out)
{
  __shared__ float sbuf[8];
  const int b = blockIdx.x;
  const bf16* qr = q + (size_t)b * Dp;
  float s = 0.f;
  for (int i = threadIdx.x; i < D; i += 256) s += (float)qr[i] * bk[i];
  s = block_sum256(s, sbuf);
  if (threadIdx.x == 0) out[b] = s;
}

// scores[b,s] = ((q@Wk)[b] . h[s,b] + q.bk) / sqrt(D)   (K-projection folded away)
__global__ __launch_bounds__(256)
void scores_kernel(const bf16* __restrict__ qk, const bf16* __restrict__ hs,
                   const float* __restrict__ qbias, int D, int Dp,
                   float* __restrict__ scores)
{
  __shared__ float sbuf[8];
  const int b = blockIdx.x / kT;
  const int s = blockIdx.x % kT;
  const bf16* hr = hs + ((size_t)s * kB + b) * Dp;
  const bf16* qr = qk + (size_t)b * Dp;
  float acc = 0.f;
  for (int i = threadIdx.x; i < D; i += 256) acc += (float)qr[i] * (float)hr[i];
  acc = block_sum256(acc, sbuf);
  if (threadIdx.x == 0)
    scores[(size_t)b * kT + s] = (acc + qbias[b]) * rsqrtf((float)D);
}

__global__ __launch_bounds__(256)
void softmax_kernel(const float* __restrict__ sc, float* __restrict__ attn)
{
  __shared__ float sbuf[8];
  const int b = blockIdx.x;
  float m = -1e30f;
  for (int s = threadIdx.x; s < kT; s += 256) m = fmaxf(m, sc[b * kT + s]);
  m = block_max256(m, sbuf);
  float sum = 0.f;
  for (int s = threadIdx.x; s < kT; s += 256) {
    float e = expf(sc[b * kT + s] - m);
    attn[b * kT + s] = e;
    sum += e;
  }
  sum = block_sum256(sum, sbuf);
  const float inv = 1.f / sum;
  for (int s = threadIdx.x; s < kT; s += 256) attn[b * kT + s] *= inv;
}

// wsum[b,n] = sum_s attn[b,s] * h[s,b,n]   (V-projection folded: applied afterwards by GEMM)
__global__ void wsum_kernel(const float* __restrict__ attn, const bf16* __restrict__ hs,
                            bf16* __restrict__ out, int Dp)
{
  const int idx = blockIdx.x * blockDim.x + threadIdx.x;
  if (idx >= kB * Dp) return;
  const int b = idx / Dp, n = idx - b * Dp;
  float acc = 0.f;
  for (int s = 0; s < kT; ++s)
    acc += attn[b * kT + s] * (float)hs[((size_t)s * kB + b) * Dp + n];
  out[idx] = (bf16)acc;
}

// cast fp32 weights -> padded bf16 [Np, Kp]; (sN,sK) strides allow transposed source.
__global__ void cast_pad_kernel(const float* __restrict__ src, bf16* __restrict__ dst,
                                int N, int K, int Kp, long sN, long sK, long total)
{
  const long idx = (long)blockIdx.x * blockDim.x + threadIdx.x;
  if (idx >= total) return;
  const int n = (int)(idx / Kp);
  const int k = (int)(idx - (long)n * Kp);
  const float v = (n < N && k < K) ? src[(long)n * sN + (long)k * sK] : 0.f;
  dst[idx] = (bf16)v;
}

// ---------------- host ----------------

extern "C" void kernel_launch(void* const* d_in, const int* in_sizes, int n_in,
                              void* d_out, int out_size, void* d_ws, size_t ws_size,
                              hipStream_t stream)
{
  (void)in_sizes; (void)n_in; (void)out_size; (void)ws_size;
  const float* x      = (const float*)d_in[0];
  const float* in_g   = (const float*)d_in[1];
  const float* in_b   = (const float*)d_in[2];
  const float* w_ih1  = (const float*)d_in[3];
  const float* w_hh1  = (const float*)d_in[4];
  const float* b_ih1  = (const float*)d_in[5];
  const float* b_hh1  = (const float*)d_in[6];
  const float* ln1_g  = (const float*)d_in[7];
  const float* ln1_b  = (const float*)d_in[8];
  const float* w_ih2  = (const float*)d_in[9];
  const float* w_hh2  = (const float*)d_in[10];
  const float* b_ih2  = (const float*)d_in[11];
  const float* b_hh2  = (const float*)d_in[12];
  const float* m1_wqkv = (const float*)d_in[13];
  const float* m1_bqkv = (const float*)d_in[14];
  const float* m1_wo   = (const float*)d_in[15];
  const float* m1_bo   = (const float*)d_in[16];
  const float* m2_wqkv = (const float*)d_in[17];
  const float* m2_bqkv = (const float*)d_in[18];
  const float* m2_wo   = (const float*)d_in[19];
  const float* m2_bo   = (const float*)d_in[20];
  const float* w_cls   = (const float*)d_in[21];
  const float* b_cls   = (const float*)d_in[22];

  char* p = (char*)d_ws;
  auto alloc = [&](size_t bytes) -> void* {
    void* r = (void*)p;
    p += (bytes + 255) & ~(size_t)255;
    return r;
  };

  // bf16 weights (padded, "W^T" form: [Nout, Kp] row-major)
  bf16* wih1b = (bf16*)alloc((size_t)kHp * kEp * 2);
  bf16* whh1b = (bf16*)alloc((size_t)kHp * kHp * 2);
  bf16* wih2b = (bf16*)alloc((size_t)kEp * kHp * 2);
  bf16* whh2b = (bf16*)alloc((size_t)kEp * kEp * 2);
  bf16* m1wq  = (bf16*)alloc((size_t)kHp * kHp * 2);
  bf16* m1wkT = (bf16*)alloc((size_t)kHp * kHp * 2);
  bf16* m1wv  = (bf16*)alloc((size_t)kHp * kHp * 2);
  bf16* m1wob = (bf16*)alloc((size_t)kHp * kHp * 2);
  bf16* m2wq  = (bf16*)alloc((size_t)kEp * kEp * 2);
  bf16* m2wkT = (bf16*)alloc((size_t)kEp * kEp * 2);
  bf16* m2wv  = (bf16*)alloc((size_t)kEp * kEp * 2);
  bf16* m2wob = (bf16*)alloc((size_t)kEp * kEp * 2);
  bf16* wclsb = (bf16*)alloc((size_t)kOp * kEp * 2);

  // activations
  bf16*  xn    = (bf16*)alloc((size_t)kT * kB * kEp * 2);
  bf16*  XW1   = (bf16*)alloc((size_t)kT * kB * kHp * 2);   // xn @ w_ih1^T (no bias)
  bf16*  h1s   = (bf16*)alloc((size_t)kT * kB * kHp * 2);
  bf16*  h2s   = (bf16*)alloc((size_t)kT * kB * kEp * 2);
  float* t1buf = (float*)alloc((size_t)kB * kHp * 4);
  float* acc1  = (float*)alloc((size_t)kT * kB * 2 * 4);
  bf16*  z1    = (bf16*)alloc((size_t)kB * kHp * 2);
  bf16*  z2    = (bf16*)alloc((size_t)kB * kEp * 2);
  bf16*  qbuf1 = (bf16*)alloc((size_t)kB * kHp * 2);
  bf16*  qkb1  = (bf16*)alloc((size_t)kB * kHp * 2);
  bf16*  wsum1 = (bf16*)alloc((size_t)kB * kHp * 2);
  bf16*  ctxb1 = (bf16*)alloc((size_t)kB * kHp * 2);
  bf16*  ctx1b = (bf16*)alloc((size_t)kB * kHp * 2);
  bf16*  qbuf2 = (bf16*)alloc((size_t)kB * kEp * 2);
  bf16*  qkb2  = (bf16*)alloc((size_t)kB * kEp * 2);
  bf16*  wsum2 = (bf16*)alloc((size_t)kB * kEp * 2);
  bf16*  ctxb2 = (bf16*)alloc((size_t)kB * kEp * 2);
  bf16*  ctx2b = (bf16*)alloc((size_t)kB * kEp * 2);
  float* qbk   = (float*)alloc(2 * kB * 4);
  float* scbuf = (float*)alloc((size_t)kB * kT * 4);
  float* atbuf = (float*)alloc((size_t)kB * kT * 4);
  bf16*  h1f   = (bf16*)alloc((size_t)kB * kHp * 2);
  bf16*  h2f   = (bf16*)alloc((size_t)kB * kEp * 2);

  auto cast = [&](const float* src, bf16* dst, int N, int K, int Np_, int Kp_, long sN, long sK) {
    long total = (long)Np_ * Kp_;
    cast_pad_kernel<<<(int)((total + 255) / 256), 256, 0, stream>>>(src, dst, N, K, Kp_, sN, sK, total);
  };
  cast(w_ih1, wih1b, kH, kE, kHp, kEp, kE, 1);
  cast(w_hh1, whh1b, kH, kH, kHp, kHp, kH, 1);
  cast(w_ih2, wih2b, kE, kH, kEp, kHp, kH, 1);
  cast(w_hh2, whh2b, kE, kE, kEp, kEp, kE, 1);
  cast(m1_wqkv,                      m1wq,  kH, kH, kHp, kHp, kH, 1);
  cast(m1_wqkv + (size_t)kH * kH,    m1wkT, kH, kH, kHp, kHp, 1, kH);  // Wk transposed (folded q@Wk)
  cast(m1_wqkv + (size_t)2 * kH * kH, m1wv, kH, kH, kHp, kHp, kH, 1);
  cast(m1_wo,                        m1wob, kH, kH, kHp, kHp, kH, 1);
  cast(m2_wqkv,                      m2wq,  kE, kE, kEp, kEp, kE, 1);
  cast(m2_wqkv + (size_t)kE * kE,    m2wkT, kE, kE, kEp, kEp, 1, kE);
  cast(m2_wqkv + (size_t)2 * kE * kE, m2wv, kE, kE, kEp, kEp, kE, 1);
  cast(m2_wo,                        m2wob, kE, kE, kEp, kEp, kE, 1);
  cast(w_cls,                        wclsb, kO, kE, kOp, kEp, kE, 1);

  hipMemsetAsync(acc1, 0, (size_t)kT * kB * 2 * 4, stream);
  hipMemsetAsync(z1, 0, (size_t)kB * kHp * 2, stream);
  hipMemsetAsync(z2, 0, (size_t)kB * kEp * 2, stream);

  input_ln_kernel<<<kT * kB, 256, 0, stream>>>(x, in_g, in_b, xn);

  auto gemm = [&](const bf16* A0, int lda0, int K0, const bf16* B0, int ldb0,
                  const bf16* A1, int lda1, int K1, const bf16* B1, int ldb1,
                  const bf16* Cin, int ldcin, const float* bias0, const float* bias1,
                  int Np_, int Nreal, int M, int ep,
                  float* Df, bf16* Db, bf16* Db2, int ldd, float* stats) {
    dim3 grid((Np_ / 16 + 3) / 4, M / 32);
    gemm_wmma_kernel<<<grid, 128, 0, stream>>>(A0, lda0, K0, B0, ldb0, A1, lda1, K1, B1, ldb1,
                                               Cin, ldcin, bias0, bias1, Np_, Nreal, ep,
                                               Df, Db, Db2, ldd, stats);
  };

  // Parallel precompute: XW1[t,b,:] = xn[t,b,:] @ w_ih1^T  (biases folded in later)
  gemm(xn, kEp, kEp, wih1b, kEp, nullptr, 0, 0, nullptr, 0, nullptr, 0, nullptr, nullptr,
       kHp, kH, kT * kB, 1, nullptr, XW1, nullptr, kHp, nullptr);

  // Sequential RNN scan
  for (int t = 0; t < kT; ++t) {
    const bf16* h1prev = t ? (h1s + (size_t)(t - 1) * kB * kHp) : z1;
    const bf16* h2prev = t ? (h2s + (size_t)(t - 1) * kB * kEp) : z2;
    // cell1: tanh(XW1[t] + b_ih1 + h1prev @ w_hh1^T + b_hh1) + LN stats
    gemm(h1prev, kHp, kHp, whh1b, kHp, nullptr, 0, 0, nullptr, 0,
         XW1 + (size_t)t * kB * kHp, kHp, b_ih1, b_hh1,
         kHp, kH, kB, 3, t1buf, nullptr, nullptr, kHp, acc1 + (size_t)t * kB * 2);
    ln_apply_kernel<<<(kB * kHp + 255) / 256, 256, 0, stream>>>(
        t1buf, acc1 + (size_t)t * kB * 2, ln1_g, ln1_b, h1s + (size_t)t * kB * kHp);
    // cell2: tanh(h1 @ w_ih2^T + b_ih2 + h2prev @ w_hh2^T + b_hh2)  (fused 2-segment K)
    gemm(h1s + (size_t)t * kB * kHp, kHp, kHp, wih2b, kHp,
         h2prev, kEp, kEp, whh2b, kEp,
         nullptr, 0, b_ih2, b_hh2,
         kEp, kE, kB, 2, nullptr, h2s + (size_t)t * kB * kEp, nullptr, kEp, nullptr);
  }

  const bf16* h1last = h1s + (size_t)(kT - 1) * kB * kHp;
  const bf16* h2last = h2s + (size_t)(kT - 1) * kB * kEp;

  // ---- MHA1 (D = H), folded: only last query row is consumed downstream ----
  gemm(h1last, kHp, kHp, m1wq, kHp, nullptr, 0, 0, nullptr, 0, nullptr, 0, m1_bqkv, nullptr,
       kHp, kH, kB, 1, nullptr, qbuf1, nullptr, kHp, nullptr);
  gemm(qbuf1, kHp, kHp, m1wkT, kHp, nullptr, 0, 0, nullptr, 0, nullptr, 0, nullptr, nullptr,
       kHp, kH, kB, 1, nullptr, qkb1, nullptr, kHp, nullptr);
  dot_qbias_kernel<<<kB, 256, 0, stream>>>(qbuf1, m1_bqkv + kH, kH, kHp, qbk);
  scores_kernel<<<kB * kT, 256, 0, stream>>>(qkb1, h1s, qbk, kH, kHp, scbuf);
  softmax_kernel<<<kB, 256, 0, stream>>>(scbuf, atbuf);
  wsum_kernel<<<(kB * kHp + 255) / 256, 256, 0, stream>>>(atbuf, h1s, wsum1, kHp);
  gemm(wsum1, kHp, kHp, m1wv, kHp, nullptr, 0, 0, nullptr, 0, nullptr, 0, m1_bqkv + 2 * kH, nullptr,
       kHp, kH, kB, 1, nullptr, ctxb1, nullptr, kHp, nullptr);
  gemm(ctxb1, kHp, kHp, m1wob, kHp, nullptr, 0, 0, nullptr, 0, nullptr, 0, m1_bo, nullptr,
       kHp, kH, kB, 1, nullptr, ctx1b, nullptr, kHp, nullptr);

  // ---- MHA2 (D = E) ----
  gemm(h2last, kEp, kEp, m2wq, kEp, nullptr, 0, 0, nullptr, 0, nullptr, 0, m2_bqkv, nullptr,
       kEp, kE, kB, 1, nullptr, qbuf2, nullptr, kEp, nullptr);
  gemm(qbuf2, kEp, kEp, m2wkT, kEp, nullptr, 0, 0, nullptr, 0, nullptr, 0, nullptr, nullptr,
       kEp, kE, kB, 1, nullptr, qkb2, nullptr, kEp, nullptr);
  dot_qbias_kernel<<<kB, 256, 0, stream>>>(qbuf2, m2_bqkv + kE, kE, kEp, qbk + kB);
  scores_kernel<<<kB * kT, 256, 0, stream>>>(qkb2, h2s, qbk + kB, kE, kEp, scbuf);
  softmax_kernel<<<kB, 256, 0, stream>>>(scbuf, atbuf);
  wsum_kernel<<<(kB * kEp + 255) / 256, 256, 0, stream>>>(atbuf, h2s, wsum2, kEp);
  gemm(wsum2, kEp, kEp, m2wv, kEp, nullptr, 0, 0, nullptr, 0, nullptr, 0, m2_bqkv + 2 * kE, nullptr,
       kEp, kE, kB, 1, nullptr, ctxb2, nullptr, kEp, nullptr);
  gemm(ctxb2, kEp, kEp, m2wob, kEp, nullptr, 0, 0, nullptr, 0, nullptr, 0, m2_bo, nullptr,
       kEp, kE, kB, 1, nullptr, ctx2b, nullptr, kEp, nullptr);

  // ---- final cells (no LN) + classifier ----
  gemm(ctx1b, kHp, kHp, whh1b, kHp, nullptr, 0, 0, nullptr, 0,
       XW1 + (size_t)(kT - 1) * kB * kHp, kHp, b_ih1, b_hh1,
       kHp, kH, kB, 2, nullptr, h1f, nullptr, kHp, nullptr);
  gemm(h1f, kHp, kHp, wih2b, kHp, ctx2b, kEp, kEp, whh2b, kEp,
       nullptr, 0, b_ih2, b_hh2,
       kEp, kE, kB, 2, nullptr, h2f, nullptr, kEp, nullptr);
  gemm(h2f, kEp, kEp, wclsb, kEp, nullptr, 0, 0, nullptr, 0, nullptr, 0, b_cls, nullptr,
       kOp, kO, kB, 0, (float*)d_out, nullptr, nullptr, kO, nullptr);
}
